// ProximityFlashAttention_54580444397663
// MI455X (gfx1250) — matrix operand order repaired and checked by reading, not compiled, not run on hardware
//
#include <hip/hip_runtime.h>

// ProximityFlashAttention for MI455X (gfx1250, wave32, WMMA).
// Pipeline: proj -> flash attention (WMMA f16 x f16 -> f32, software-pipelined)
// -> output proj.  Distance-squared bias is folded into 5 extra contraction
// dims so the entire logit computation runs on the matrix pipe.

static constexpr int H_    = 4;
static constexpr int CH_   = 16;
static constexpr int AD_   = 64;
static constexpr int VD_   = 16;
static constexpr int N_    = 4096;
static constexpr int KDIM  = 96;    // 64 features + 5 pos dims + 27 zero pad

typedef _Float16 v16h __attribute__((ext_vector_type(16)));
typedef _Float16 v8h  __attribute__((ext_vector_type(8)));
typedef float    v8f  __attribute__((ext_vector_type(8)));

// Load a 16-half A/B operand fragment from two 16-byte chunks (b128 pair).
__device__ __forceinline__ v16h ld_frag(const _Float16* __restrict__ p)
{
    v8h lo = *(const v8h*)(p);
    v8h hi = *(const v8h*)(p + 16);
    return __builtin_shufflevector(lo, hi, 0,1,2,3,4,5,6,7,8,9,10,11,12,13,14,15);
}

// ---------------------------------------------------------------------------
// Kernel 1: projections.  Produces
//   Qe[h][n][96] (f16, activated, pos-augmented)
//   Ke[h][n][96] (f16, activated, pos-augmented)
//   Vt[h][64][n] (f16, transposed V for contiguous WMMA A-operand loads)
// block = 256 threads = 16 tokens x 16 channels, grid = (N/16, H)
// ---------------------------------------------------------------------------
__global__ __launch_bounds__(256)
void pfa_proj_kernel(const float* __restrict__ ax, const float* __restrict__ vx,
                     const float* __restrict__ pos_k, const float* __restrict__ pos_q,
                     const float* __restrict__ W_aq, const float* __restrict__ W_vq,
                     const float* __restrict__ W_ak, const float* __restrict__ W_vk,
                     const float* __restrict__ W_av, const float* __restrict__ W_vv,
                     _Float16* __restrict__ Qe, _Float16* __restrict__ Ke,
                     _Float16* __restrict__ Vt)
{
    __shared__ float s_ax[16 * AD_];
    __shared__ float s_vx[16 * VD_ * 3];

    const int h  = blockIdx.y;
    const int n0 = blockIdx.x * 16;
    const int tid = threadIdx.x;

    for (int i = tid; i < 16 * AD_; i += 256)       s_ax[i] = ax[(size_t)n0 * AD_ + i];
    for (int i = tid; i < 16 * VD_ * 3; i += 256)   s_vx[i] = vx[(size_t)n0 * VD_ * 3 + i];
    __syncthreads();

    const int t = tid >> 4;        // token in tile
    const int c = tid & 15;        // channel
    const int n = n0 + t;

    const float* waq = W_aq + ((size_t)h * CH_ + c) * AD_;
    const float* wak = W_ak + ((size_t)h * CH_ + c) * AD_;
    const float* wav = W_av + ((size_t)h * CH_ + c) * AD_;
    const float* wvq = W_vq + ((size_t)h * CH_ + c) * VD_;
    const float* wvk = W_vk + ((size_t)h * CH_ + c) * VD_;
    const float* wvv = W_vv + ((size_t)h * CH_ + c) * VD_;

    float qa = 0.f, ka = 0.f, va = 0.f;
    #pragma unroll 8
    for (int j = 0; j < AD_; ++j) {
        float x = s_ax[t * AD_ + j];
        qa = fmaf(waq[j], x, qa);
        ka = fmaf(wak[j], x, ka);
        va = fmaf(wav[j], x, va);
    }
    float qv[3] = {0.f, 0.f, 0.f}, kv[3] = {0.f, 0.f, 0.f}, vv[3] = {0.f, 0.f, 0.f};
    #pragma unroll
    for (int j = 0; j < VD_; ++j) {
        float wq = wvq[j], wk = wvk[j], wvw = wvv[j];
        #pragma unroll
        for (int p = 0; p < 3; ++p) {
            float x = s_vx[(t * VD_ + j) * 3 + p];
            qv[p] = fmaf(wq,  x, qv[p]);
            kv[p] = fmaf(wk,  x, kv[p]);
            vv[p] = fmaf(wvw, x, vv[p]);
        }
    }

    // vector activation: g * (1+|g|^2)^(-1/4), group = (a, v0, v1, v2)
    float qn = qa*qa + qv[0]*qv[0] + qv[1]*qv[1] + qv[2]*qv[2];
    float kn = ka*ka + kv[0]*kv[0] + kv[1]*kv[1] + kv[2]*kv[2];
    float qs = sqrtf(rsqrtf(1.0f + qn));   // (1+n)^(-0.25)
    float ks = sqrtf(rsqrtf(1.0f + kn));

    const size_t row = ((size_t)h * N_ + n) * KDIM + 4 * c;
    Qe[row + 0] = (_Float16)(qa * qs);
    Qe[row + 1] = (_Float16)(qv[0] * qs);
    Qe[row + 2] = (_Float16)(qv[1] * qs);
    Qe[row + 3] = (_Float16)(qv[2] * qs);
    Ke[row + 0] = (_Float16)(ka * ks);
    Ke[row + 1] = (_Float16)(kv[0] * ks);
    Ke[row + 2] = (_Float16)(kv[1] * ks);
    Ke[row + 3] = (_Float16)(kv[2] * ks);

    // V transposed: Vt[h][feat][n]
    const size_t vcol = (size_t)h * AD_ * N_ + n;
    Vt[vcol + (size_t)(4 * c + 0) * N_] = (_Float16)va;
    Vt[vcol + (size_t)(4 * c + 1) * N_] = (_Float16)vv[0];
    Vt[vcol + (size_t)(4 * c + 2) * N_] = (_Float16)vv[1];
    Vt[vcol + (size_t)(4 * c + 3) * N_] = (_Float16)vv[2];

    // position-augmentation dims 64..95 (one thread per token)
    if (c == 0) {
        const float invr = 1.0f / (float)(1 << (2 * h));   // R0SQ = 1,4,16,64
        const float* pq = pos_q + ((size_t)h * N_ + n) * 3;
        const float* pk = pos_k + ((size_t)h * N_ + n) * 3;
        float pqx = pq[0], pqy = pq[1], pqz = pq[2];
        float pkx = pk[0], pky = pk[1], pkz = pk[2];
        _Float16* qp = Qe + ((size_t)h * N_ + n) * KDIM + AD_;
        _Float16* kp = Ke + ((size_t)h * N_ + n) * KDIM + AD_;
        qp[0] = (_Float16)(2.0f * pqx * invr);
        qp[1] = (_Float16)(2.0f * pqy * invr);
        qp[2] = (_Float16)(2.0f * pqz * invr);
        qp[3] = (_Float16)((pqx*pqx + pqy*pqy + pqz*pqz) * invr);
        qp[4] = (_Float16)1.0f;
        kp[0] = (_Float16)pkx;
        kp[1] = (_Float16)pky;
        kp[2] = (_Float16)pkz;
        kp[3] = (_Float16)(-1.0f);
        kp[4] = (_Float16)(-(pkx*pkx + pky*pky + pkz*pkz) * invr);
        #pragma unroll
        for (int k = 5; k < 32; ++k) { qp[k] = (_Float16)0.0f; kp[k] = (_Float16)0.0f; }
    }
}

// ---------------------------------------------------------------------------
// Kernel 2: flash attention, one 16-query tile per wave, 32 keys / iteration,
// software-pipelined with register double-buffering of the Ke A-tiles.
//   S^T = Ke (A) x Qe^T (B)   : 6 wmma (k = 96)
//   O^T = V^T (A) x P^T (B)   : 4 wmma (k = 32, 64 output features)
// grid = (N/64, H), block = 128 (4 waves)
// ---------------------------------------------------------------------------
__global__ __launch_bounds__(128)
void pfa_attn_kernel(const _Float16* __restrict__ Qe, const _Float16* __restrict__ Ke,
                     const _Float16* __restrict__ Vt, float* __restrict__ O)
{
    const int h    = blockIdx.y;
    const int wave = threadIdx.x >> 5;
    const int lane = threadIdx.x & 31;
    const int col  = lane & 15;        // query (B/C column) or feature row
    const int hi   = lane >> 4;        // upper half-wave flag
    const int qb   = (blockIdx.x * 4 + wave) * 16;

    const _Float16* Qh = Qe + (size_t)h * N_ * KDIM;
    const _Float16* Kh = Ke + (size_t)h * N_ * KDIM;
    const _Float16* Vh = Vt + (size_t)h * AD_ * N_;

    // B operand (Qe^T), resident: lane column = query qb+col,
    // lower half-wave holds k rows 0..7,16..23 of each 32-chunk, upper holds 8..15,24..31.
    v16h bq[3];
    {
        const _Float16* qrow = Qh + (size_t)(qb + col) * KDIM;
        #pragma unroll
        for (int kc = 0; kc < 3; ++kc)
            bq[kc] = ld_frag(qrow + kc * 32 + hi * 8);
    }

    v8f acc[4] = {v8f{}, v8f{}, v8f{}, v8f{}};   // O^T tiles (feat x query)
    float m = -1e30f, l = 0.0f;

    // Ke A-tile loader: [subtile (16 keys)][k-chunk]
    auto load_ak = [&](v16h (&ak)[2][3], int jb) {
        #pragma unroll
        for (int kc = 0; kc < 3; ++kc) {
            ak[0][kc] = ld_frag(Kh + (size_t)(jb + col)      * KDIM + kc * 32 + hi * 8);
            ak[1][kc] = ld_frag(Kh + (size_t)(jb + 16 + col) * KDIM + kc * 32 + hi * 8);
        }
    };

    // One 32-key step: consume `cur`, preload `nxt` for key tile `jn`.
    auto step = [&](v16h (&cur)[2][3], int jb, v16h (&nxt)[2][3], int jn) {
        // V^T A-tiles for this step: issue now, consumed after softmax.
        v16h av[4];
        #pragma unroll
        for (int f = 0; f < 4; ++f)
            av[f] = *(const v16h*)(Vh + (size_t)(16 * f + col) * N_ + jb + hi * 16);
        // Preload next Ke tile while this step computes.
        if (jn < N_) load_ak(nxt, jn);

        // ---- S^T = Ke * Qe^T ----
        v8f s0 = {}, s1 = {};
        #pragma unroll
        for (int kc = 0; kc < 3; ++kc) {
            s0 = __builtin_amdgcn_wmma_f32_16x16x32_f16(false, cur[0][kc], false, bq[kc],
                                                        (short)0, s0, false, false);
            s1 = __builtin_amdgcn_wmma_f32_16x16x32_f16(false, cur[1][kc], false, bq[kc],
                                                        (short)0, s1, false, false);
        }

        // ---- online softmax (lane column == query; combine across halves) ----
        float cmax = -1e30f;
        #pragma unroll
        for (int v = 0; v < 8; ++v) cmax = fmaxf(cmax, fmaxf(s0[v], s1[v]));
        cmax = fmaxf(cmax, __shfl_xor(cmax, 16, 32));
        float mn    = fmaxf(m, cmax);
        float alpha = __expf(m - mn);
        m = mn;

        float p0[8], p1[8], csum = 0.0f;
        #pragma unroll
        for (int v = 0; v < 8; ++v) {
            p0[v] = __expf(s0[v] - mn);
            p1[v] = __expf(s1[v] - mn);
            csum += p0[v] + p1[v];
        }
        csum += __shfl_xor(csum, 16, 32);
        l = l * alpha + csum;

        #pragma unroll
        for (int f = 0; f < 4; ++f)
            #pragma unroll
            for (int v = 0; v < 8; ++v) acc[f][v] *= alpha;

        // ---- build P^T B-operand: C-layout -> B-layout is a half-wave swap ----
        float x0[8], x1[8];
        #pragma unroll
        for (int v = 0; v < 8; ++v) {
            x0[v] = __shfl_xor(p0[v], 16, 32);
            x1[v] = __shfl_xor(p1[v], 16, 32);
        }
        v16h pb;
        #pragma unroll
        for (int v = 0; v < 4; ++v) {          // key rows j = 2v, 2v+1 (+16 upper half)
            float e0 = hi ? x1[2 * v]     : p0[2 * v];
            float e1 = hi ? x1[2 * v + 1] : p0[2 * v + 1];
            pb[2 * v]     = (_Float16)e0;
            pb[2 * v + 1] = (_Float16)e1;
        }
        #pragma unroll
        for (int v = 4; v < 8; ++v) {
            int w = 2 * v - 8;
            float e0 = hi ? p1[w]     : x0[w];
            float e1 = hi ? p1[w + 1] : x0[w + 1];
            pb[2 * v]     = (_Float16)e0;
            pb[2 * v + 1] = (_Float16)e1;
        }

        // ---- O^T += V^T * P^T : 4 feature tiles ----
        #pragma unroll
        for (int f = 0; f < 4; ++f)
            acc[f] = __builtin_amdgcn_wmma_f32_16x16x32_f16(false, av[f], false, pb,
                                                            (short)0, acc[f], false, false);
    };

    // N/32 = 128 steps, unrolled x2 so the double buffers rotate for free.
    v16h akA[2][3], akB[2][3];
    load_ak(akA, 0);
    #pragma unroll 1
    for (int jb = 0; jb < N_; jb += 64) {
        step(akA, jb,      akB, jb + 32);
        step(akB, jb + 32, akA, jb + 64);
    }

    // ---- epilogue: O[h][q][feat] = acc / (l * sqrt(N)) ----
    const float inv = __fdividef(1.0f, l * sqrtf((float)N_));
    const size_t obase = ((size_t)h * N_ + (qb + col)) * AD_;
    #pragma unroll
    for (int f = 0; f < 4; ++f) {
        v8f t;
        #pragma unroll
        for (int v = 0; v < 8; ++v) t[v] = acc[f][v] * inv;
        *(v8f*)(O + obase + 16 * f + hi * 8) = t;   // 8 consecutive features
    }
}

// ---------------------------------------------------------------------------
// Kernel 3: output projections summed over heads.
// block = 128 = 2 tokens x 64 output dims, grid = N/2
// ---------------------------------------------------------------------------
__global__ __launch_bounds__(128)
void pfa_outproj_kernel(const float* __restrict__ O, const float* __restrict__ W_ao,
                        const float* __restrict__ W_vo, float* __restrict__ a_out,
                        float* __restrict__ v_out)
{
    const int tid = threadIdx.x;
    const int n = blockIdx.x * 2 + (tid >> 6);
    const int d = tid & 63;

    float accA = 0.0f;
    #pragma unroll
    for (int h = 0; h < H_; ++h) {
        const float* o = O + ((size_t)h * N_ + n) * AD_;
        const float* w = W_ao + ((size_t)h * AD_ + d) * CH_;
        #pragma unroll
        for (int i = 0; i < CH_; ++i) accA = fmaf(w[i], o[i], accA);
    }
    a_out[(size_t)n * AD_ + d] = accA;

    if (d < VD_ * 3) {
        const int j = d / 3, p = d % 3;
        float accV = 0.0f;
        #pragma unroll
        for (int h = 0; h < H_; ++h) {
            const float* o = O + ((size_t)h * N_ + n) * AD_;
            const float* w = W_vo + ((size_t)h * VD_ + j) * CH_;
            #pragma unroll
            for (int i = 0; i < CH_; ++i) accV = fmaf(w[i], o[CH_ + 3 * i + p], accV);
        }
        v_out[((size_t)n * VD_ + j) * 3 + p] = accV;
    }
}

// ---------------------------------------------------------------------------
extern "C" void kernel_launch(void* const* d_in, const int* in_sizes, int n_in,
                              void* d_out, int out_size, void* d_ws, size_t ws_size,
                              hipStream_t stream)
{
    const float* ax    = (const float*)d_in[0];
    const float* vx    = (const float*)d_in[1];
    const float* pos_k = (const float*)d_in[2];
    const float* pos_q = (const float*)d_in[3];
    const float* W_aq  = (const float*)d_in[4];
    const float* W_vq  = (const float*)d_in[5];
    const float* W_ak  = (const float*)d_in[6];
    const float* W_vk  = (const float*)d_in[7];
    const float* W_av  = (const float*)d_in[8];
    const float* W_vv  = (const float*)d_in[9];
    const float* W_ao  = (const float*)d_in[10];
    const float* W_vo  = (const float*)d_in[11];

    // workspace layout (bytes)
    const size_t QE = (size_t)H_ * N_ * KDIM * sizeof(_Float16);   // 3 MiB
    const size_t VT = (size_t)H_ * AD_ * N_ * sizeof(_Float16);    // 2 MiB
    char* ws = (char*)d_ws;
    _Float16* Qe = (_Float16*)(ws);
    _Float16* Ke = (_Float16*)(ws + QE);
    _Float16* Vt = (_Float16*)(ws + 2 * QE);
    float*    Ob = (float*)   (ws + 2 * QE + VT);                  // 4 MiB

    float* a_out = (float*)d_out;
    float* v_out = a_out + (size_t)N_ * AD_;

    pfa_proj_kernel<<<dim3(N_ / 16, H_), 256, 0, stream>>>(
        ax, vx, pos_k, pos_q, W_aq, W_vq, W_ak, W_vk, W_av, W_vv, Qe, Ke, Vt);

    pfa_attn_kernel<<<dim3(N_ / 64, H_), 128, 0, stream>>>(Qe, Ke, Vt, Ob);

    pfa_outproj_kernel<<<dim3(N_ / 2), 128, 0, stream>>>(Ob, W_ao, W_vo, a_out, v_out);
}
